// GINNet_7584912244763
// MI455X (gfx1250) — matrix operand order repaired and checked by reading, not compiled, not run on hardware
//
#include <hip/hip_runtime.h>
#include <hip/hip_bf16.h>

typedef __attribute__((ext_vector_type(16))) __bf16 v16bf;
typedef __attribute__((ext_vector_type(8)))  float  v8f;

#define N_NODES  100000
#define N_EDGES  1600000
#define IN_DIM   64
#define HIDDEN   128
#define N_GRAPHS 512

// ---------------------------------------------------------------------------
// Edge scatter-add: agg[dst][d] += x[src][d].  One lane per (edge, dim); a
// wave32 covers 32 aligned dims of ONE edge (DIM is 64/128), so the edge id is
// wave-uniform -> readfirstlane turns the two index reads into scalar loads.
// Gather stream may be bf16 (halved bytes); accumulation stays fp32 atomics.
// ---------------------------------------------------------------------------
template<int LOGD, typename T>
__global__ __launch_bounds__(256) void scatter_add_kernel(
    const T* __restrict__ x, const int* __restrict__ ei,
    float* __restrict__ agg, int total)
{
    int gid = blockIdx.x * 256 + threadIdx.x;
    if (gid >= total) return;
    int e = __builtin_amdgcn_readfirstlane(gid >> LOGD);   // wave-uniform
    int d = gid & ((1 << LOGD) - 1);
    int s = ei[e];
    int t = ei[N_EDGES + e];
    atomicAdd(&agg[(t << LOGD) + d], (float)x[(s << LOGD) + d]);
}

// ---------------------------------------------------------------------------
// WMMA fragment loads from LDS (bf16), per ISA 7.12.2 layouts.
// A 16x32: lane M=lane&15; half 0 -> K {0..7,16..23}, half 1 -> K {8..15,24..31}
// B 32x16: lane N=lane&15; half 0 -> K {0..7,16..23}, half 1 -> K {8..15,24..31}
// Row stride is padded by 8 bf16 (16 B) -> conflict-free ds_load_b128.
// ---------------------------------------------------------------------------
__device__ __forceinline__ v16bf load_a_frag(const __bf16* rowp, int kb, int half) {
    union { v16bf v; uint4 q[2]; } u;
    u.q[0] = *(const uint4*)(rowp + kb + half * 8);
    u.q[1] = *(const uint4*)(rowp + kb + 16 + half * 8);
    return u.v;
}
__device__ __forceinline__ v16bf load_b_frag(const __bf16* rowp, int kb, int half) {
    union { v16bf v; uint4 q[2]; } u;
    const __bf16* p = rowp + kb + half * 8;
    u.q[0] = *(const uint4*)(p);
    u.q[1] = *(const uint4*)(p + 16);
    return u.v;
}

// ---------------------------------------------------------------------------
// Fused GIN MLP tile kernel: out = [relu]( relu((x+agg) @ Wa + ba) @ Wb + bb )
// 256 threads = 8 waves. Block tile: 128 rows x 128 cols. Wave w owns rows
// [16w,16w+16); each wave computes 8 16x16 WMMA tiles across N=128.
// H1 tile lives only in LDS (bf16) between the two GEMMs.
// TIN/TOUT select fp32 or bf16 global activation streams.
// ---------------------------------------------------------------------------
template<int K_IN, typename TIN, typename TOUT, bool RELU_OUT>
__global__ __launch_bounds__(256) void gin_layer_kernel(
    const TIN*  __restrict__ x,  const float* __restrict__ agg,
    const float* __restrict__ Wa, const float* __restrict__ ba,
    const float* __restrict__ Wb, const float* __restrict__ bb,
    TOUT* __restrict__ out, int n_nodes)
{
    constexpr int KPA = K_IN + 8;     // padded bf16 row stride (A / Wa^T)
    constexpr int KPH = HIDDEN + 8;   // padded bf16 row stride (H1 / Wb^T)

    extern __shared__ __align__(16) unsigned char smem_raw[];
    __bf16* sA  = (__bf16*)smem_raw;        // [128][KPA]  A = x + agg
    __bf16* sWa = sA  + 128 * KPA;          // [128][KPA]  Wa transposed [n][k]
    __bf16* sH  = sWa + 128 * KPA;          // [128][KPH]  H1 tile
    __bf16* sWb = sH  + 128 * KPH;          // [128][KPH]  Wb transposed [n][k]

    const int tid  = threadIdx.x;
    const int wave = tid >> 5;
    const int lane = tid & 31;
    const int half = lane >> 4;
    const int cl   = lane & 15;
    const int row0 = blockIdx.x * 128;

    // Stage weights transposed (k-major rows per output column)
    for (int i = tid; i < K_IN * HIDDEN; i += 256) {
        int k = i / HIDDEN, n = i % HIDDEN;
        sWa[n * KPA + k] = (__bf16)Wa[i];
    }
    for (int i = tid; i < HIDDEN * HIDDEN; i += 256) {
        int k = i / HIDDEN, n = i % HIDDEN;
        sWb[n * KPH + k] = (__bf16)Wb[i];
    }
    // Stage A = x + agg as bf16, zero-fill tail rows
    for (int i = tid; i < 128 * K_IN; i += 256) {
        int r = i / K_IN, k = i % K_IN;
        int gr = row0 + r;
        float v = 0.0f;
        if (gr < n_nodes) v = (float)x[(size_t)gr * K_IN + k] + agg[(size_t)gr * K_IN + k];
        sA[r * KPA + k] = (__bf16)v;
    }
    __syncthreads();

    const v8f vzero = {0.f, 0.f, 0.f, 0.f, 0.f, 0.f, 0.f, 0.f};
    v8f acc[8];
#pragma unroll
    for (int t = 0; t < 8; ++t) acc[t] = vzero;

    // GEMM 1: (128 x K_IN) @ (K_IN x 128)
    const __bf16* aRow = sA + (16 * wave + cl) * KPA;
#pragma unroll
    for (int kb = 0; kb < K_IN; kb += 32) {
        v16bf a = load_a_frag(aRow, kb, half);
#pragma unroll
        for (int t = 0; t < 8; ++t) {
            v16bf b = load_b_frag(sWa + (16 * t + cl) * KPA, kb, half);
            acc[t] = __builtin_amdgcn_wmma_f32_16x16x32_bf16(
                false, a, false, b, (short)0, acc[t], false, false);
        }
    }

    // Epilogue 1: bias + ReLU -> H1 tile (bf16 in LDS)
    // C/D layout: VGPR j holds row (j + 8*half), col = cl (within 16x16 tile)
#pragma unroll
    for (int t = 0; t < 8; ++t) {
        int col = 16 * t + cl;
        float bias = ba[col];
#pragma unroll
        for (int j = 0; j < 8; ++j) {
            int r = 16 * wave + 8 * half + j;
            float v = acc[t][j] + bias;
            v = fmaxf(v, 0.0f);
            sH[r * KPH + col] = (__bf16)v;
        }
    }
    __syncthreads();

    // GEMM 2: (128 x 128) @ (128 x 128)
#pragma unroll
    for (int t = 0; t < 8; ++t) acc[t] = vzero;
    const __bf16* hRow = sH + (16 * wave + cl) * KPH;
#pragma unroll
    for (int kb = 0; kb < HIDDEN; kb += 32) {
        v16bf a = load_a_frag(hRow, kb, half);
#pragma unroll
        for (int t = 0; t < 8; ++t) {
            v16bf b = load_b_frag(sWb + (16 * t + cl) * KPH, kb, half);
            acc[t] = __builtin_amdgcn_wmma_f32_16x16x32_bf16(
                false, a, false, b, (short)0, acc[t], false, false);
        }
    }

    // Epilogue 2: bias (+ outer ReLU) -> global (fp32 or bf16)
#pragma unroll
    for (int t = 0; t < 8; ++t) {
        int col = 16 * t + cl;
        float bias = bb[col];
#pragma unroll
        for (int j = 0; j < 8; ++j) {
            int gr = row0 + 16 * wave + 8 * half + j;
            if (gr < n_nodes) {
                float v = acc[t][j] + bias;
                if (RELU_OUT) v = fmaxf(v, 0.0f);
                out[(size_t)gr * HIDDEN + col] = (TOUT)v;
            }
        }
    }
}

// ---------------------------------------------------------------------------
// Global mean pool, stage 1: per-graph sums + counts via L2-resident atomics.
// ---------------------------------------------------------------------------
__global__ __launch_bounds__(256) void pool_sum_kernel(
    const float* __restrict__ h, const int* __restrict__ batch,
    float* __restrict__ sums, float* __restrict__ counts)
{
    int gid = blockIdx.x * 256 + threadIdx.x;
    if (gid >= N_NODES * HIDDEN) return;
    int node = gid >> 7;
    int d    = gid & 127;
    int g    = batch[node];
    atomicAdd(&sums[g * HIDDEN + d], h[gid]);
    if (d == 0) atomicAdd(&counts[g], 1.0f);
}

// Stage 2: one wave per graph: out[g] = dot(sums[g], Wc)/max(cnt,1) + bc
__global__ __launch_bounds__(32) void pool_out_kernel(
    const float* __restrict__ sums, const float* __restrict__ counts,
    const float* __restrict__ Wc, const float* __restrict__ bc,
    float* __restrict__ out)
{
    int g = blockIdx.x;
    int l = threadIdx.x;
    float cnt = fmaxf(counts[g], 1.0f);
    float acc = 0.0f;
#pragma unroll
    for (int d = l; d < HIDDEN; d += 32) acc += sums[g * HIDDEN + d] * Wc[d];
#pragma unroll
    for (int off = 16; off > 0; off >>= 1) acc += __shfl_xor(acc, off, 32);
    if (l == 0) out[g] = acc / cnt + bc[0];
}

// ---------------------------------------------------------------------------
extern "C" void kernel_launch(void* const* d_in, const int* in_sizes, int n_in,
                              void* d_out, int out_size, void* d_ws, size_t ws_size,
                              hipStream_t stream) {
    const float* x    = (const float*)d_in[0];
    const int*   ei   = (const int*)d_in[1];   // edge_index [2, N_EDGES]
    const int*   batch= (const int*)d_in[2];
    const float* W1a  = (const float*)d_in[3];
    const float* b1a  = (const float*)d_in[4];
    const float* W1b  = (const float*)d_in[5];
    const float* b1b  = (const float*)d_in[6];
    const float* W2a  = (const float*)d_in[7];
    const float* b2a  = (const float*)d_in[8];
    const float* W2b  = (const float*)d_in[9];
    const float* b2b  = (const float*)d_in[10];
    const float* Wc   = (const float*)d_in[11];
    const float* bc   = (const float*)d_in[12];
    float* out = (float*)d_out;

    // Workspace layout
    float*  agg    = (float*)d_ws;                         // N_NODES*HIDDEN fp32 (reused)
    float*  h2     = agg    + (size_t)N_NODES * HIDDEN;    // N_NODES*HIDDEN fp32
    float*  sums   = h2     + (size_t)N_NODES * HIDDEN;    // N_GRAPHS*HIDDEN fp32
    float*  counts = sums   + (size_t)N_GRAPHS * HIDDEN;   // N_GRAPHS fp32
    __bf16* h1     = (__bf16*)(counts + N_GRAPHS);         // N_NODES*HIDDEN bf16

    const int gemm_blocks = (N_NODES + 127) / 128;
    const size_t shmem64  = (size_t)(2 * 128 * (IN_DIM + 8) + 2 * 128 * (HIDDEN + 8)) * 2;
    const size_t shmem128 = (size_t)(4 * 128 * (HIDDEN + 8)) * 2;

    // ---- Layer 1 ----
    hipMemsetAsync(agg, 0, (size_t)N_NODES * IN_DIM * sizeof(float), stream);
    {
        int total = N_EDGES * IN_DIM;
        scatter_add_kernel<6, float><<<(total + 255) / 256, 256, 0, stream>>>(x, ei, agg, total);
    }
    gin_layer_kernel<IN_DIM, float, __bf16, true><<<gemm_blocks, 256, shmem64, stream>>>(
        x, agg, W1a, b1a, W1b, b1b, h1, N_NODES);

    // ---- Layer 2 ----
    hipMemsetAsync(agg, 0, (size_t)N_NODES * HIDDEN * sizeof(float), stream);
    {
        int total = N_EDGES * HIDDEN;
        scatter_add_kernel<7, __bf16><<<(total + 255) / 256, 256, 0, stream>>>(h1, ei, agg, total);
    }
    gin_layer_kernel<HIDDEN, __bf16, float, true><<<gemm_blocks, 256, shmem128, stream>>>(
        h1, agg, W2a, b2a, W2b, b2b, h2, N_NODES);

    // ---- Mean pool + head ----
    hipMemsetAsync(sums, 0, ((size_t)N_GRAPHS * HIDDEN + N_GRAPHS) * sizeof(float), stream);
    pool_sum_kernel<<<(N_NODES * HIDDEN + 255) / 256, 256, 0, stream>>>(h2, batch, sums, counts);
    pool_out_kernel<<<N_GRAPHS, 32, 0, stream>>>(sums, counts, Wc, bc, out);
}
